// MolformerModel_24824910971287
// MI455X (gfx1250) — matrix operand order repaired and checked by reading, not compiled, not run on hardware
//
#include <hip/hip_runtime.h>
#include <hip/hip_bf16.h>
#include <math.h>

typedef __attribute__((ext_vector_type(16))) __bf16 v16bf;
typedef __attribute__((ext_vector_type(8)))  __bf16 v8bf;
typedef __attribute__((ext_vector_type(8)))  float  v8f;
typedef __attribute__((ext_vector_type(4)))  unsigned u32x4;
typedef __attribute__((ext_vector_type(8)))  unsigned u32x8;

#define Bb 64
#define Ss 512
#define Dd 768
#define NLl 12
#define Hh 12
#define Dh 64
#define NFf 32
#define PHIphi 64
#define Mm (Bb*Ss)           // 32768

#define TILE_M 128
#define TILE_N 128
#define TILE_K 32
#define LDSP  40             // padded row stride (elements): 32 data + 8 pad

// ---------- helpers ----------
__device__ __forceinline__ v8f v8f_zero() {
  v8f r;
#pragma unroll
  for (int i = 0; i < 8; ++i) r[i] = 0.0f;
  return r;
}
__device__ __forceinline__ v16bf cat8(v8bf lo, v8bf hi) {
  return __builtin_shufflevector(lo, hi, 0,1,2,3,4,5,6,7,8,9,10,11,12,13,14,15);
}
// CDNA5 async copy: 16B global -> LDS, tracked by ASYNCcnt
__device__ __forceinline__ void async_copy_b128(unsigned lds_off, const void* gptr) {
  asm volatile("global_load_async_to_lds_b128 %0, %1, off"
               :: "v"(lds_off), "v"(gptr) : "memory");
}
__device__ __forceinline__ unsigned lds_addr32(const void* p) {
  return (unsigned)(unsigned long long)p;   // low 32 bits of flat addr == LDS offset
}
__device__ __forceinline__ unsigned rfl(unsigned v) {
  return (unsigned)__builtin_amdgcn_readfirstlane((int)v);
}

// CDNA5 Tensor Data Mover: 2D tile (tile_dim0=32 elems of 2B, tile_dim1=128 rows,
// row stride = 768 elems) global -> LDS, with HW padding of 4 DWORDs every
// 16 DWORDs so LDS rows land on the LDSP=40-element padded layout.
// Tracked by TENSORcnt.
__device__ __forceinline__ void tdm_load_tile(unsigned lds_off, const void* gptr,
                                              unsigned tensor_rows) {
  unsigned long long ga = (unsigned long long)gptr;
  u32x4 g0;
  g0[0] = 1u;                                      // count=1, user descriptor
  g0[1] = rfl(lds_off);                            // lds_addr (bits 63:32)
  g0[2] = rfl((unsigned)ga);                       // global_addr[31:0]
  g0[3] = rfl((unsigned)(ga >> 32)) | 0x80000000u; // global_addr[56:32] | type=2
  u32x8 g1;
  // word0: workgroup_mask=0 | data_size=1(2B)<<16 | pad_enable<<20
  //        | pad_interval=3(16 DW)<<22 | pad_amount=3(4 DW)<<25
  g1[0] = 0x06D10000u;
  g1[1] = ((unsigned)Dd) << 16;                    // tensor_dim0 = 768 (low16)
  g1[2] = rfl(tensor_rows) << 16;                  // tensor_dim0 hi=0 | tensor_dim1 low16
  g1[3] = 32u << 16;                               // tensor_dim1 hi=0 | tile_dim0 = 32
  g1[4] = 128u;                                    // tile_dim1 = 128, tile_dim2 = 0
  g1[5] = (unsigned)Dd;                            // tensor_dim0_stride = 768 (low32)
  g1[6] = 0u;
  g1[7] = 0u;
  u32x4 g2; g2[0] = g2[1] = g2[2] = g2[3] = 0u;
  u32x4 g3; g3[0] = g3[1] = g3[2] = g3[3] = 0u;
  asm volatile("tensor_load_to_lds %0, %1, %2, %3"
               :: "s"(g0), "s"(g1), "s"(g2), "s"(g3) : "memory");
}

// ---------- weight convert + transpose: W (K x N) fp32 -> WT (N x K) bf16 ----------
__global__ __launch_bounds__(256) void k_wcvt(const float* __restrict__ W,
                                              __bf16* __restrict__ WT) {
  __shared__ float tile[32][33];
  int bid = blockIdx.x;                 // NL * 24 * 24 blocks
  int l  = bid / 576;
  int r  = bid % 576;
  int tk = r / 24, tn = r % 24;
  const float* src = W + (size_t)l * Dd * Dd;
  __bf16* dst = WT + (size_t)l * Dd * Dd;
  int tx = threadIdx.x & 31, ty = threadIdx.x >> 5;
#pragma unroll
  for (int i = 0; i < 32; i += 8)
    tile[ty + i][tx] = src[(size_t)(tk*32 + ty + i) * Dd + tn*32 + tx];
  __syncthreads();
#pragma unroll
  for (int i = 0; i < 32; i += 8)
    dst[(size_t)(tn*32 + ty + i) * Dd + tk*32 + tx] = (__bf16)tile[tx][ty + i];
}

// ---------- embedding gather ----------
__global__ __launch_bounds__(256) void k_embed(const int* __restrict__ ids,
                                               const float* __restrict__ emb,
                                               float* __restrict__ x,
                                               __bf16* __restrict__ xb) {
  size_t i = (size_t)blockIdx.x * 256 + threadIdx.x;   // M*D total
  int tok = (int)(i / Dd);
  int d   = (int)(i % Dd);
  float v = emb[(size_t)ids[tok] * Dd + d];
  x[i]  = v;
  xb[i] = (__bf16)v;
}

// ---------- main bf16 WMMA GEMM: C = A(MxK) * WT(NxK)^T + bias ----------
// A tile: per-lane async-to-LDS.  B (weight) tile: TDM tensor_load_to_lds.
// EPI 0: fp32 out.  EPI 1: bf16 out with exact GELU.
template<int EPI>
__global__ __launch_bounds__(256) void k_gemm(const __bf16* __restrict__ A,
                                              const __bf16* __restrict__ BT,
                                              const float*  __restrict__ bias,
                                              float*  __restrict__ Cf,
                                              __bf16* __restrict__ Cb,
                                              int Mdim, int Ndim, int Kdim) {
  __shared__ __align__(16) __bf16 As[2][TILE_M][LDSP];
  __shared__ __align__(16) __bf16 Bs[2][TILE_N][LDSP];

  const int tid  = threadIdx.x;
  const int lane = tid & 31;
  const int wave = tid >> 5;
  const int wm   = wave & 1;    // 64-row slab
  const int wn   = wave >> 1;   // 32-col slab

  const int nTilesN = Ndim / TILE_N;
  const int m0 = ((int)blockIdx.x / nTilesN) * TILE_M;
  const int n0 = ((int)blockIdx.x % nTilesN) * TILE_N;
  const int T  = Kdim / TILE_K;

  __builtin_prefetch(A + (size_t)m0 * Kdim, 0, 0);   // global_prefetch_b8
  __builtin_prefetch(BT + (size_t)n0 * Kdim, 0, 0);

  auto issue = [&](int t, int buf) {
    const long long kbase = (long long)t * TILE_K;
#pragma unroll
    for (int i = 0; i < 2; ++i) {
      int c = i * 256 + tid;          // 512 x 16B chunks for the A tile
      int row = c >> 2, kc = c & 3;
      const __bf16* ga = A + (long long)(m0 + row) * Kdim + kbase + kc * 8;
      async_copy_b128(lds_addr32(&As[buf][row][kc * 8]), ga);
    }
    if (tid < 32) {                   // one TDM issue per block (wave 0)
      const __bf16* gb = BT + (long long)n0 * Kdim + kbase;
      tdm_load_tile(lds_addr32(&Bs[buf][0][0]), gb, (unsigned)Ndim);
    }
  };

  v8f acc[4][2];
#pragma unroll
  for (int mf = 0; mf < 4; ++mf)
#pragma unroll
    for (int nf = 0; nf < 2; ++nf) acc[mf][nf] = v8f_zero();

  issue(0, 0);

  const int arow_base = wm * 64 + (lane & 15);
  const int kA0 = (lane >> 4) << 4;     // 0 or 16 (A: 16 contiguous K)
  const int bcol_base = wn * 32 + (lane & 15);
  const int kB0 = (lane >> 4) << 4;     // 0 or 16 (B: 16 contiguous K)

  for (int t = 0; t < T; ++t) {
    const int buf = t & 1;
    if (t + 1 < T) {
      issue(t + 1, buf ^ 1);
      asm volatile("s_wait_asynccnt 2" ::: "memory");   // this stage's 2 A-chunks done
      __builtin_amdgcn_s_wait_tensorcnt(1);             // this stage's TDM done
    } else {
      asm volatile("s_wait_asynccnt 0" ::: "memory");
      __builtin_amdgcn_s_wait_tensorcnt(0);
    }
    __syncthreads();   // propagate TDM/async completion to all waves

    v16bf afrag[4], bfrag[2];
#pragma unroll
    for (int mf = 0; mf < 4; ++mf) {
      const __bf16* p = &As[buf][arow_base + mf * 16][0];
      v8bf lo = *(const v8bf*)(p + kA0);
      v8bf hi = *(const v8bf*)(p + kA0 + 8);
      afrag[mf] = cat8(lo, hi);
    }
#pragma unroll
    for (int nf = 0; nf < 2; ++nf) {
      const __bf16* p = &Bs[buf][bcol_base + nf * 16][kB0];
      v8bf lo = *(const v8bf*)(p);
      v8bf hi = *(const v8bf*)(p + 8);
      bfrag[nf] = cat8(lo, hi);
    }
#pragma unroll
    for (int mf = 0; mf < 4; ++mf)
#pragma unroll
      for (int nf = 0; nf < 2; ++nf)
        acc[mf][nf] = __builtin_amdgcn_wmma_f32_16x16x32_bf16(
            false, afrag[mf], false, bfrag[nf], (short)0, acc[mf][nf], false, false);
    __syncthreads();
  }

  // epilogue
  const int rsel = (lane >> 4) << 3;    // rows +0 or +8 within 16x16 frag
#pragma unroll
  for (int mf = 0; mf < 4; ++mf) {
#pragma unroll
    for (int nf = 0; nf < 2; ++nf) {
      const int col  = n0 + wn * 32 + nf * 16 + (lane & 15);
      const float bv = bias[col];
      const int rowb = m0 + wm * 64 + mf * 16 + rsel;
#pragma unroll
      for (int vr = 0; vr < 8; ++vr) {
        float val = acc[mf][nf][vr] + bv;
        long long idx = (long long)(rowb + vr) * Ndim + col;
        if constexpr (EPI == 0) {
          Cf[idx] = val;
        } else {
          float g = 0.5f * val * (1.0f + erff(val * 0.70710678f));
          Cb[idx] = (__bf16)g;
        }
      }
    }
  }
}

// ---------- RoPE + random-feature map (in place over q,k) ----------
__global__ __launch_bounds__(256) void k_features(float* __restrict__ q,
                                                  float* __restrict__ k,
                                                  const float* __restrict__ omega_l,
                                                  const int* __restrict__ amask) {
  __shared__ float qs[Dd], ks[Dd], qr[Dd], kr[Dd];
  __shared__ float om[Dh * NFf];
  const int m = blockIdx.x;
  const int b = m >> 9, s = m & 511;
  const int t = threadIdx.x;
  for (int i = t; i < Dh * NFf; i += 256) om[i] = omega_l[i];
  for (int i = t; i < Dd; i += 256) {
    qs[i] = q[(size_t)m * Dd + i];
    ks[i] = k[(size_t)m * Dd + i];
  }
  __syncthreads();
  for (int i = t; i < Dd; i += 256) {
    int dl = i & 63, j = dl & 31;
    float ang = (float)s * __expf(-0.28782313662425572f * (float)j); // 10000^(-j/32)
    float sn, c;
    __sincosf(ang, &sn, &c);
    float rq = (dl < 32) ? -qs[i + 32] : qs[i - 32];
    float rk = (dl < 32) ? -ks[i + 32] : ks[i - 32];
    qr[i] = qs[i] * c + rq * sn;
    kr[i] = ks[i] * c + rk * sn;
  }
  __syncthreads();
  const float maskf = (float)amask[(b << 9) + s];
  const float phi_scale = 0.17677669529663687f;   // sqrt(2/PHI)
  for (int o = t; o < Dd; o += 256) {
    int h = o >> 6, c = o & 63, f = c & 31;
    const float* qh = &qr[h << 6];
    const float* kh = &kr[h << 6];
    float uq = 0.f, uk = 0.f;
#pragma unroll 8
    for (int d = 0; d < Dh; ++d) {
      float w = om[d * NFf + f];
      uq += qh[d] * w;
      uk += kh[d] * w;
    }
    float Qv = ((c < 32) ? __cosf(uq) : __sinf(uq)) * phi_scale;
    float Kv = ((c < 32) ? __cosf(uk) : __sinf(uk)) * phi_scale * maskf;
    q[(size_t)m * Dd + o] = Qv;
    k[(size_t)m * Dd + o] = Kv;
  }
}

// ---------- per (b,h): KV = K^T V  (PHI x Dh), ksum = sum_s K ----------
__global__ __launch_bounds__(256) void k_kv(const float* __restrict__ kf,
                                            const float* __restrict__ v,
                                            float* __restrict__ kv,
                                            float* __restrict__ ksum) {
  __shared__ float kk[4][64], vv4[4][64];
  const int bh = blockIdx.x;
  const int b = bh / Hh, h = bh % Hh;
  const int t = threadIdx.x;
  const int f = t >> 2, d0 = (t & 3) << 4;
  float acc[16];
#pragma unroll
  for (int i = 0; i < 16; ++i) acc[i] = 0.f;
  float sk = 0.f;
  for (int s0 = 0; s0 < Ss; s0 += 4) {
    int ss = t >> 6, ln = t & 63;
    size_t base = ((size_t)(b * Ss + s0 + ss)) * Dd + h * Dh + ln;
    kk[ss][ln]  = kf[base];
    vv4[ss][ln] = v[base];
    __syncthreads();
#pragma unroll
    for (int q4 = 0; q4 < 4; ++q4) {
      float kvv = kk[q4][f];
      if ((t & 3) == 0) sk += kvv;
#pragma unroll
      for (int i = 0; i < 16; ++i) acc[i] += kvv * vv4[q4][d0 + i];
    }
    __syncthreads();
  }
  size_t obase = ((size_t)bh * PHIphi + f) * Dh + d0;
#pragma unroll
  for (int i = 0; i < 16; ++i) kv[obase + i] = acc[i];
  if ((t & 3) == 0) ksum[(size_t)bh * PHIphi + f] = sk;
}

// ---------- attention combine: att = (Q . KV) * Z -> bf16 ----------
__global__ __launch_bounds__(256) void k_attn(const float* __restrict__ qf,
                                              const float* __restrict__ kv,
                                              const float* __restrict__ ksum,
                                              __bf16* __restrict__ ab) {
  __shared__ float Qs[Dd];
  __shared__ float z[Hh];
  const int m = blockIdx.x;
  const int b = m >> 9;
  const int t = threadIdx.x;
  for (int i = t; i < Dd; i += 256) Qs[i] = qf[(size_t)m * Dd + i];
  __syncthreads();
  if (t < Hh) {
    const float* ks_ = &ksum[((size_t)b * Hh + t) * PHIphi];
    float a = 0.f;
#pragma unroll 8
    for (int f = 0; f < PHIphi; ++f) a += Qs[t * Dh + f] * ks_[f];
    z[t] = 1.0f / (a + 1e-6f);
  }
  __syncthreads();
  for (int o = t; o < Dd; o += 256) {
    int h = o >> 6, d = o & 63;
    const float* kvh = &kv[(((size_t)b * Hh + h) * PHIphi) * Dh + d];
    const float* qh  = &Qs[h << 6];
    float a = 0.f;
#pragma unroll 8
    for (int f = 0; f < PHIphi; ++f) a += qh[f] * kvh[(size_t)f * Dh];
    ab[(size_t)m * Dd + o] = (__bf16)(a * z[h]);
  }
}

// ---------- residual add + LayerNorm; emits fp32 + bf16 ----------
__global__ __launch_bounds__(256) void k_add_ln(const float* __restrict__ xin,
                                                const float* __restrict__ delta,
                                                const float* __restrict__ g,
                                                const float* __restrict__ bln,
                                                float* __restrict__ xout,
                                                __bf16* __restrict__ xbout) {
  __shared__ float row[Dd];
  __shared__ float red[256];
  const int m = blockIdx.x;
  const int t = threadIdx.x;
  float lsum = 0.f;
  for (int i = t; i < Dd; i += 256) {
    float v = xin[(size_t)m * Dd + i] + delta[(size_t)m * Dd + i];
    row[i] = v;
    lsum += v;
  }
  red[t] = lsum;
  __syncthreads();
  for (int s = 128; s > 0; s >>= 1) {
    if (t < s) red[t] += red[t + s];
    __syncthreads();
  }
  const float mean = red[0] / (float)Dd;
  __syncthreads();
  float lsq = 0.f;
  for (int i = t; i < Dd; i += 256) {
    float dq = row[i] - mean;
    lsq += dq * dq;
  }
  red[t] = lsq;
  __syncthreads();
  for (int s = 128; s > 0; s >>= 1) {
    if (t < s) red[t] += red[t + s];
    __syncthreads();
  }
  const float rstd = rsqrtf(red[0] / (float)Dd + 1e-5f);
  for (int i = t; i < Dd; i += 256) {
    float v = (row[i] - mean) * rstd * g[i] + bln[i];
    xout[(size_t)m * Dd + i]  = v;
    xbout[(size_t)m * Dd + i] = (__bf16)v;
  }
}

// ---------- outputs ----------
__global__ __launch_bounds__(256) void k_copy_x0(const float* __restrict__ x,
                                                 float* __restrict__ out) {
  int i = blockIdx.x * 256 + threadIdx.x;   // 512*768 elems
  out[i] = x[i];
}
__global__ __launch_bounds__(256) void k_pool(const float* __restrict__ x,
                                              const int* __restrict__ amask,
                                              float* __restrict__ out) {
  const int b = blockIdx.x / 3;
  const int d = (blockIdx.x % 3) * 256 + threadIdx.x;
  float sum = 0.f, cnt = 0.f;
  for (int s = 0; s < Ss; ++s) {
    float mf = (float)amask[b * Ss + s];
    sum += x[((size_t)(b * Ss + s)) * Dd + d] * mf;
    cnt += mf;
  }
  out[Ss * Dd + b * Dd + d] = sum / fmaxf(cnt, 1e-9f);
}

// ---------- host orchestration ----------
extern "C" void kernel_launch(void* const* d_in, const int* in_sizes, int n_in,
                              void* d_out, int out_size, void* d_ws, size_t ws_size,
                              hipStream_t stream) {
  (void)in_sizes; (void)n_in; (void)out_size; (void)ws_size;
  const int*   ids   = (const int*)d_in[0];
  const int*   amask = (const int*)d_in[1];
  const float* emb   = (const float*)d_in[2];
  const float* Wq = (const float*)d_in[3];  const float* bq = (const float*)d_in[4];
  const float* Wk = (const float*)d_in[5];  const float* bk = (const float*)d_in[6];
  const float* Wv = (const float*)d_in[7];  const float* bv = (const float*)d_in[8];
  const float* Wo = (const float*)d_in[9];  const float* bo = (const float*)d_in[10];
  const float* omega = (const float*)d_in[11];
  const float* ln1g = (const float*)d_in[12]; const float* ln1b = (const float*)d_in[13];
  const float* ln2g = (const float*)d_in[14]; const float* ln2b = (const float*)d_in[15];
  const float* W1 = (const float*)d_in[16]; const float* b1 = (const float*)d_in[17];
  const float* W2 = (const float*)d_in[18]; const float* b2 = (const float*)d_in[19];
  float* out = (float*)d_out;

  // workspace layout
  char* p = (char*)d_ws;
  const size_t wBytes = (size_t)NLl * Dd * Dd * sizeof(__bf16);   // per weight tensor
  __bf16* wqT = (__bf16*)p;               p += wBytes;
  __bf16* wkT = (__bf16*)p;               p += wBytes;
  __bf16* wvT = (__bf16*)p;               p += wBytes;
  __bf16* woT = (__bf16*)p;               p += wBytes;
  __bf16* w1T = (__bf16*)p;               p += wBytes;
  __bf16* w2T = (__bf16*)p;               p += wBytes;
  const size_t actF = (size_t)Mm * Dd * sizeof(float);
  float* x   = (float*)p;                 p += actF;
  float* qb  = (float*)p;                 p += actF;
  float* kb  = (float*)p;                 p += actF;
  float* vb  = (float*)p;                 p += actF;
  const size_t actB = (size_t)Mm * Dd * sizeof(__bf16);
  __bf16* xb = (__bf16*)p;                p += actB;
  __bf16* ab = (__bf16*)p;                p += actB;
  float* kvb = (float*)p;                 p += (size_t)Bb * Hh * PHIphi * Dh * sizeof(float);
  float* ksb = (float*)p;                 p += (size_t)Bb * Hh * PHIphi * sizeof(float);

  const int cvtBlocks  = NLl * 24 * 24;          // 6912 per tensor
  const int gemmBlocks = (Mm / TILE_M) * (Dd / TILE_N);  // 1536
  const size_t lstep = (size_t)Dd * Dd;          // 589824 elems per layer matrix

  // convert + transpose all weights to bf16 (N x K)
  k_wcvt<<<cvtBlocks, 256, 0, stream>>>(Wq, wqT);
  k_wcvt<<<cvtBlocks, 256, 0, stream>>>(Wk, wkT);
  k_wcvt<<<cvtBlocks, 256, 0, stream>>>(Wv, wvT);
  k_wcvt<<<cvtBlocks, 256, 0, stream>>>(Wo, woT);
  k_wcvt<<<cvtBlocks, 256, 0, stream>>>(W1, w1T);
  k_wcvt<<<cvtBlocks, 256, 0, stream>>>(W2, w2T);

  // embedding
  k_embed<<<(Mm * Dd) / 256, 256, 0, stream>>>(ids, emb, x, xb);

  for (int l = 0; l < NLl; ++l) {
    // q/k/v projections (bf16 WMMA, fp32 out)
    k_gemm<0><<<gemmBlocks, 256, 0, stream>>>(xb, wqT + (size_t)l * lstep, bq + l * Dd,
                                              qb, (__bf16*)nullptr, Mm, Dd, Dd);
    k_gemm<0><<<gemmBlocks, 256, 0, stream>>>(xb, wkT + (size_t)l * lstep, bk + l * Dd,
                                              kb, (__bf16*)nullptr, Mm, Dd, Dd);
    k_gemm<0><<<gemmBlocks, 256, 0, stream>>>(xb, wvT + (size_t)l * lstep, bv + l * Dd,
                                              vb, (__bf16*)nullptr, Mm, Dd, Dd);
    // RoPE + feature map (in place), masked K
    k_features<<<Mm, 256, 0, stream>>>(qb, kb, omega + (size_t)l * Dh * NFf, amask);
    // KV and ksum per (b,h)
    k_kv<<<Bb * Hh, 256, 0, stream>>>(kb, vb, kvb, ksb);
    // combine -> bf16 attention output
    k_attn<<<Mm, 256, 0, stream>>>(qb, kvb, ksb, ab);
    // output projection
    k_gemm<0><<<gemmBlocks, 256, 0, stream>>>(ab, woT + (size_t)l * lstep, bo + l * Dd,
                                              qb, (__bf16*)nullptr, Mm, Dd, Dd);
    // x = LN1(x + attn_out)
    k_add_ln<<<Mm, 256, 0, stream>>>(x, qb, ln1g + l * Dd, ln1b + l * Dd, x, xb);
    // FFN: GELU GEMM writes bf16 directly
    k_gemm<1><<<gemmBlocks, 256, 0, stream>>>(xb, w1T + (size_t)l * lstep, b1 + l * Dd,
                                              (float*)nullptr, ab, Mm, Dd, Dd);
    k_gemm<0><<<gemmBlocks, 256, 0, stream>>>(ab, w2T + (size_t)l * lstep, b2 + l * Dd,
                                              qb, (__bf16*)nullptr, Mm, Dd, Dd);
    // x = LN2(x + y)
    k_add_ln<<<Mm, 256, 0, stream>>>(x, qb, ln2g + l * Dd, ln2b + l * Dd, x, xb);
  }

  // outputs: x[0] then masked mean pool
  k_copy_x0<<<(Ss * Dd) / 256, 256, 0, stream>>>(x, out);
  k_pool<<<Bb * 3, 256, 0, stream>>>(x, amask, out);
}
